// MixtureOfExperts_73624329388231
// MI455X (gfx1250) — compile-verified, matrix operand order checked by reading, not run on hardware
//
#include <hip/hip_runtime.h>
#include <cstdint>
#include <cstddef>

// ---------------------------------------------------------------------------
// MoE: gate GEMM via split-f16 WMMA (fp32-accurate scores) -> top2 route ->
//      selected-expert GEMM (f16 WMMA, f32 acc, 32-row tiles) -> LayerNorm ->
//      GELU -> weighted atomic combine.
// Shapes: B=8192, D_IN=1024, gate hidden=512, E=8, TOP_K=2, OUT=1024
// ---------------------------------------------------------------------------

#define BB   8192
#define DIN  1024
#define DG   512
#define NE   8
#define DOUT 1024
#define ROWS_PER_TILE 32
#define MAX_TILES (BB / ROWS_PER_TILE)   // 256: worst case all rows -> 1 expert

typedef _Float16 f16;
typedef __attribute__((ext_vector_type(16))) _Float16 v16h;
typedef __attribute__((ext_vector_type(8)))  _Float16 v8h;
typedef __attribute__((ext_vector_type(8)))  float    v8f;

__device__ __forceinline__ float gelu_exact(float v) {
    return 0.5f * v * (1.0f + erff(v * 0.70710678118654752440f));
}

// Load a 16-element f16 fragment (A 16x32 or B 32x16, contiguous along K).
// CDNA5 ISA 7.12.2: lanes 0-15 hold K=0..7 (elems 0-7) and K=16..23
// (elems 8-15); lanes 16-31 hold K=8..15 and K=24..31.
__device__ __forceinline__ v16h load_frag16(const f16* __restrict__ p, int lane) {
    const bool hi = lane >= 16;
    v8h lo = *(const v8h*)(p + (hi ? 8 : 0));
    v8h hh = *(const v8h*)(p + (hi ? 24 : 16));
    v16h r;
#pragma unroll
    for (int i = 0; i < 8; ++i) { r[i] = lo[i]; r[i + 8] = hh[i]; }
    return r;
}

// ---------------------------------------------------------------------------
// 1a) Split conversion for the gate path (and expert A-matrix):
//     xhi = f16(x); xlo = f16(x - f32(xhi))           [hi+lo ~ 22-bit mantissa]
//     w1thi/w1tlo: gate_w1 transposed to [o][k], split the same way.
// ---------------------------------------------------------------------------
__global__ void cvt1_kernel(const float* __restrict__ x,
                            const float* __restrict__ w1,
                            f16* __restrict__ xhi, f16* __restrict__ xlo,
                            f16* __restrict__ w1thi, f16* __restrict__ w1tlo) {
    const int64_t tid    = (int64_t)blockIdx.x * blockDim.x + threadIdx.x;
    const int64_t stride = (int64_t)gridDim.x * blockDim.x;

    for (int64_t i = tid; i < (int64_t)BB * DIN; i += stride) {
        float v = x[i];
        f16 h = (f16)v;
        xhi[i] = h;
        xlo[i] = (f16)(v - (float)h);
    }
    for (int64_t i = tid; i < (int64_t)DG * DIN; i += stride) {
        int o = (int)(i >> 10);
        int k = (int)(i & (DIN - 1));
        float v = w1[(int64_t)k * DG + o];
        f16 h = (f16)v;
        w1thi[i] = h;
        w1tlo[i] = (f16)(v - (float)h);
    }
}

// 1b) Expert weights: wet[e][o][k] = f16(We[e][k][o]); runs after routing so
//     its buffer can alias xlo (dead by then).
__global__ void cvt2_kernel(const float* __restrict__ We, f16* __restrict__ wet) {
    const int64_t tid    = (int64_t)blockIdx.x * blockDim.x + threadIdx.x;
    const int64_t stride = (int64_t)gridDim.x * blockDim.x;
    for (int64_t i = tid; i < (int64_t)NE * DOUT * DIN; i += stride) {
        int e   = (int)(i >> 20);
        int rem = (int)(i & ((1 << 20) - 1));
        int o   = rem >> 10;
        int k   = rem & (DIN - 1);
        wet[i] = (f16)We[((int64_t)e << 20) + ((int64_t)k << 10) + o];
    }
}

// ---------------------------------------------------------------------------
// 2) Gate GEMM, fp32-accurate via 3x f16 WMMA per K-chunk:
//    x@w ~= xh@wh + xh@wl + xl@wh   (err ~2^-22, preserves fp32 top-k)
//    g = GELU(x @ gate_w1 + b1) stored fp32.
// ---------------------------------------------------------------------------
__global__ void __launch_bounds__(256)
gate1_kernel(const f16* __restrict__ xhi, const f16* __restrict__ xlo,
             const f16* __restrict__ w1thi, const f16* __restrict__ w1tlo,
             const float* __restrict__ b1, float* __restrict__ g) {
    const int wave = threadIdx.x >> 5;
    const int lane = threadIdx.x & 31;
    const int m15  = lane & 15;

    const int rowTile = blockIdx.x >> 2;        // 512 row tiles
    const int colGrp  = blockIdx.x & 3;         // 4 groups of 128 cols
    const int colBase = colGrp * 128 + wave * 16;

    const f16* ah_p = xhi   + (size_t)(rowTile * 16 + m15) * DIN;
    const f16* al_p = xlo   + (size_t)(rowTile * 16 + m15) * DIN;
    const f16* bh_p = w1thi + (size_t)(colBase + m15) * DIN;
    const f16* bl_p = w1tlo + (size_t)(colBase + m15) * DIN;

    v8f acc = {};
    for (int k0 = 0; k0 < DIN; k0 += 32) {
        v16h ah = load_frag16(ah_p + k0, lane);
        v16h al = load_frag16(al_p + k0, lane);
        v16h bh = load_frag16(bh_p + k0, lane);
        v16h bl = load_frag16(bl_p + k0, lane);
        acc = __builtin_amdgcn_wmma_f32_16x16x32_f16(false, ah, false, bh, (short)0, acc, false, false);
        acc = __builtin_amdgcn_wmma_f32_16x16x32_f16(false, ah, false, bl, (short)0, acc, false, false);
        acc = __builtin_amdgcn_wmma_f32_16x16x32_f16(false, al, false, bh, (short)0, acc, false, false);
    }

    const int col  = colBase + m15;
    const float bs = b1[col];
#pragma unroll
    for (int m = 0; m < 8; ++m) {
        int r = rowTile * 16 + m + ((lane >> 4) << 3);
        g[(size_t)r * DG + col] = gelu_exact(acc[m] + bs);
    }
}

// ---------------------------------------------------------------------------
// 3) Routing: scores = g @ w2 + b2 (serial fp32, deterministic);
//    first-occurrence top-2 (jax tie rule); 2-way softmax; compact lists.
// ---------------------------------------------------------------------------
__global__ void route_kernel(const float* __restrict__ g,
                             const float* __restrict__ w2,
                             const float* __restrict__ b2,
                             int* __restrict__ rowlist,
                             float* __restrict__ roww,
                             int* __restrict__ counts) {
    int b = blockIdx.x * blockDim.x + threadIdx.x;
    if (b >= BB) return;

    float acc[NE];
#pragma unroll
    for (int e = 0; e < NE; ++e) acc[e] = b2[e];

    const float* gr = g + (size_t)b * DG;
    for (int k = 0; k < DG; ++k) {
        float gv = gr[k];
        const float* wr = w2 + (size_t)k * NE;
#pragma unroll
        for (int e = 0; e < NE; ++e) acc[e] += gv * wr[e];
    }

    int i1 = 0;
#pragma unroll
    for (int e = 1; e < NE; ++e) if (acc[e] > acc[i1]) i1 = e;
    int i2 = -1;
#pragma unroll
    for (int e = 0; e < NE; ++e)
        if (e != i1 && (i2 < 0 || acc[e] > acc[i2])) i2 = e;

    float ex = expf(acc[i2] - acc[i1]);          // <= 1, no overflow
    float wa = 1.0f / (1.0f + ex);
    float wb = ex * wa;

    int p1 = atomicAdd(&counts[i1], 1);
    rowlist[i1 * BB + p1] = b;  roww[i1 * BB + p1] = wa;
    int p2 = atomicAdd(&counts[i2], 1);
    rowlist[i2 * BB + p2] = b;  roww[i2 * BB + p2] = wb;
}

// ---------------------------------------------------------------------------
// 4) Expert kernel: block = (expert e, tile of 32 routed rows).
//    512 threads = 16 waves; wave -> 4 col-tiles x 2 row-groups: each B
//    fragment feeds two WMMAs (halves L2 traffic on We vs 16-row tiles).
//    Then bias + cross-wave LayerNorm + exact GELU + routed atomicAdd.
// ---------------------------------------------------------------------------
__global__ void __launch_bounds__(512)
expert_kernel(const f16* __restrict__ xh, const f16* __restrict__ wet,
              const float* __restrict__ be, const float* __restrict__ lng,
              const float* __restrict__ lnb,
              const int* __restrict__ rowlist, const float* __restrict__ roww,
              const int* __restrict__ counts, float* __restrict__ out) {
    const int e = blockIdx.x >> 8;              // / MAX_TILES
    const int t = blockIdx.x & (MAX_TILES - 1);
    const int cnt = counts[e];
    const int nt  = (cnt + ROWS_PER_TILE - 1) >> 5;
    if (t >= nt) return;

    __shared__ int   sRow[32];
    __shared__ float sW[32];
    __shared__ float sS[16][32];
    __shared__ float sQ[16][32];
    __shared__ float sMu[32];
    __shared__ float sRs[32];

    if (threadIdx.x < 32) {
        int idx = t * 32 + (int)threadIdx.x;
        int cl  = idx < cnt ? idx : cnt - 1;     // clamp tail (weight 0)
        sRow[threadIdx.x] = rowlist[e * BB + cl];
        sW[threadIdx.x]   = (idx < cnt) ? roww[e * BB + cl] : 0.0f;
    }
    __syncthreads();

    const int wave = threadIdx.x >> 5;           // 0..15
    const int lane = threadIdx.x & 31;
    const int m15  = lane & 15;
    const int half = lane >> 4;

    const f16* xrow0 = xh + (size_t)sRow[m15]      * DIN;
    const f16* xrow1 = xh + (size_t)sRow[16 + m15] * DIN;
    const f16* wbase = wet + ((size_t)e << 20);

    v8f acc0[4], acc1[4];
#pragma unroll
    for (int j = 0; j < 4; ++j) {
        acc0[j] = (v8f){0.f,0.f,0.f,0.f,0.f,0.f,0.f,0.f};
        acc1[j] = (v8f){0.f,0.f,0.f,0.f,0.f,0.f,0.f,0.f};
    }

    for (int k0 = 0; k0 < DIN; k0 += 32) {
        v16h a0 = load_frag16(xrow0 + k0, lane);
        v16h a1 = load_frag16(xrow1 + k0, lane);
#pragma unroll
        for (int j = 0; j < 4; ++j) {
            int col = wave * 64 + j * 16 + m15;
            v16h bf = load_frag16(wbase + (size_t)col * DIN + k0, lane);
            acc0[j] = __builtin_amdgcn_wmma_f32_16x16x32_f16(
                false, a0, false, bf, (short)0, acc0[j], false, false);
            acc1[j] = __builtin_amdgcn_wmma_f32_16x16x32_f16(
                false, a1, false, bf, (short)0, acc1[j], false, false);
        }
    }

    // bias + per-row partial stats over this wave's 64 columns
    float bias[4], lgv[4], lbv[4];
#pragma unroll
    for (int j = 0; j < 4; ++j) {
        int col = wave * 64 + j * 16 + m15;
        bias[j] = be [e * DOUT + col];
        lgv[j]  = lng[e * DOUT + col];
        lbv[j]  = lnb[e * DOUT + col];
    }

    float s0[8], q0[8], s1[8], q1[8];
#pragma unroll
    for (int m = 0; m < 8; ++m) { s0[m]=0.f; q0[m]=0.f; s1[m]=0.f; q1[m]=0.f; }
#pragma unroll
    for (int j = 0; j < 4; ++j)
#pragma unroll
        for (int m = 0; m < 8; ++m) {
            float h0 = acc0[j][m] + bias[j];
            float h1 = acc1[j][m] + bias[j];
            acc0[j][m] = h0;  acc1[j][m] = h1;
            s0[m] += h0;  q0[m] += h0 * h0;
            s1[m] += h1;  q1[m] += h1 * h1;
        }

    // reduce across the 16 lanes of each half
#pragma unroll
    for (int off = 1; off < 16; off <<= 1)
#pragma unroll
        for (int m = 0; m < 8; ++m) {
            s0[m] += __shfl_xor(s0[m], off, 32);
            q0[m] += __shfl_xor(q0[m], off, 32);
            s1[m] += __shfl_xor(s1[m], off, 32);
            q1[m] += __shfl_xor(q1[m], off, 32);
        }
    if (m15 == 0) {
#pragma unroll
        for (int m = 0; m < 8; ++m) {
            int r = m + 8 * half;
            sS[wave][r]      = s0[m];  sQ[wave][r]      = q0[m];
            sS[wave][16 + r] = s1[m];  sQ[wave][16 + r] = q1[m];
        }
    }
    __syncthreads();

    if (threadIdx.x < 32) {
        float S = 0.f, Q = 0.f;
#pragma unroll
        for (int w = 0; w < 16; ++w) { S += sS[w][threadIdx.x]; Q += sQ[w][threadIdx.x]; }
        float mu  = S * (1.0f / DOUT);
        float var = Q * (1.0f / DOUT) - mu * mu;
        sMu[threadIdx.x] = mu;
        sRs[threadIdx.x] = rsqrtf(var + 1e-5f);
    }
    __syncthreads();

    // normalize, GELU, weighted accumulate (two row groups)
#pragma unroll
    for (int grp = 0; grp < 2; ++grp) {
#pragma unroll
        for (int m = 0; m < 8; ++m) {
            int   r  = grp * 16 + m + 8 * half;
            float w  = sW[r];
            if (w == 0.0f) continue;
            float mu = sMu[r], rs = sRs[r];
            size_t orow = (size_t)sRow[r] * DOUT;
#pragma unroll
            for (int j = 0; j < 4; ++j) {
                int col = wave * 64 + j * 16 + m15;
                float h = (grp == 0) ? acc0[j][m] : acc1[j][m];
                float y = (h - mu) * rs * lgv[j] + lbv[j];
                y = gelu_exact(y);
                atomicAdd(&out[orow + col], w * y);
            }
        }
    }
}

// ---------------------------------------------------------------------------
extern "C" void kernel_launch(void* const* d_in, const int* in_sizes, int n_in,
                              void* d_out, int out_size, void* d_ws, size_t ws_size,
                              hipStream_t stream) {
    const float* x   = (const float*)d_in[0];
    const float* w1  = (const float*)d_in[1];
    const float* b1  = (const float*)d_in[2];
    const float* w2  = (const float*)d_in[3];
    const float* b2  = (const float*)d_in[4];
    const float* We  = (const float*)d_in[5];
    const float* be  = (const float*)d_in[6];
    const float* lng = (const float*)d_in[7];
    const float* lnb = (const float*)d_in[8];
    float* out = (float*)d_out;
    (void)in_sizes; (void)n_in;

    char*  ws = (char*)d_ws;
    size_t o  = 0;
    auto carve = [&](size_t bytes) -> char* {
        char* p = ws + o;
        o += (bytes + 255) & ~(size_t)255;
        return p;
    };
    f16*   xhi     = (f16*)  carve((size_t)BB * DIN * sizeof(f16));
    // xlo is dead after gate1; wet (written by cvt2 after routing) aliases it.
    f16*   xlo     = (f16*)  carve((size_t)NE * DOUT * DIN * sizeof(f16));
    f16*   wet     = xlo;
    float* gbuf    = (float*)carve((size_t)BB * DG * sizeof(float));
    f16*   w1thi   = (f16*)  carve((size_t)DG * DIN * sizeof(f16));
    f16*   w1tlo   = (f16*)  carve((size_t)DG * DIN * sizeof(f16));
    int*   rowlist = (int*)  carve((size_t)NE * BB * sizeof(int));
    float* roww    = (float*)carve((size_t)NE * BB * sizeof(float));
    int*   counts  = (int*)  carve(256);
    if (o > ws_size) return;   // workspace too small: bail (deterministic)

    hipMemsetAsync(counts, 0, 256, stream);
    hipMemsetAsync(out, 0, (size_t)out_size * sizeof(float), stream);

    cvt1_kernel <<<2048, 256, 0, stream>>>(x, w1, xhi, xlo, w1thi, w1tlo);
    gate1_kernel<<<(BB / 16) * 4, 256, 0, stream>>>(xhi, xlo, w1thi, w1tlo, b1, gbuf);
    route_kernel<<<BB / 256, 256, 0, stream>>>(gbuf, w2, b2, rowlist, roww, counts);
    cvt2_kernel <<<2048, 256, 0, stream>>>(We, wet);   // after gate1: aliases xlo
    expert_kernel<<<NE * MAX_TILES, 512, 0, stream>>>(xhi, wet, be, lng, lnb,
                                                      rowlist, roww, counts, out);
}
